// GammaContrastReconLoss_27556510171593
// MI455X (gfx1250) — matrix operand order repaired and checked by reading, not compiled, not run on hardware
//
#include <hip/hip_runtime.h>
#include <math.h>

// ---- constants from the reference ----
#define GAMMA_C   0.5f
#define TEMP_C    0.1f
#define ZVP_C     0.01f
#define ZNP_C     0.001f
#define EPS_C     1e-8f

#define B_ROWS    1024      // batch
#define C_COLS    4096      // feature dim
#define N_ROWS    2048      // 2*B
#define RECON_N   67108864  // 1024*512*128
#define RECON_N4  16777216  // /4

typedef __attribute__((ext_vector_type(4)))  _Float16 v4h;
typedef __attribute__((ext_vector_type(8)))  _Float16 v8h;
typedef __attribute__((ext_vector_type(16))) _Float16 v16h;
typedef __attribute__((ext_vector_type(8)))  float    v8f;

// workspace layout (floats):
// ws[0]=recon_sum  ws[1]=znorm_sum  ws[2]=zvar_sum  ws[3]=pad
// ws[4            .. 4+2048)  rowExpSum[2048]
// ws[4+2048       .. 4+4096)  pos[2048]
// ws[4+4096       .. 4+6144)  invnorm[2048]
#define WS_FLOATS (4 + 3*2048)

// ---------------- init: zero the accumulator region ----------------
__global__ void init_ws_kernel(float* ws) {
    int i = blockIdx.x * blockDim.x + threadIdx.x;
    if (i < WS_FLOATS) ws[i] = 0.0f;
}

// ---------------- recon: sum((m*y - t)^2), float4 streaming (HBM-bound term) ----------------
__global__ __launch_bounds__(256) void recon_kernel(const float4* __restrict__ y,
                                                    const float4* __restrict__ t,
                                                    const float4* __restrict__ m,
                                                    float* __restrict__ ws) {
    float acc = 0.0f;
    int stride = gridDim.x * blockDim.x;
    for (int i = blockIdx.x * blockDim.x + threadIdx.x; i < RECON_N4; i += stride) {
        float4 yv = y[i], tv = t[i], mv = m[i];
        float dx = mv.x * yv.x - tv.x;
        float dy = mv.y * yv.y - tv.y;
        float dz = mv.z * yv.z - tv.z;
        float dw = mv.w * yv.w - tv.w;
        acc += dx*dx + dy*dy + dz*dz + dw*dw;
    }
    __shared__ float red[256];
    red[threadIdx.x] = acc;
    __syncthreads();
    for (int s = 128; s > 0; s >>= 1) {
        if (threadIdx.x < s) red[threadIdx.x] += red[threadIdx.x + s];
        __syncthreads();
    }
    if (threadIdx.x == 0) atomicAdd(&ws[0], red[0]);
}

// ---------------- per-row L2 norms of z = [zi; zj]; Σ norm over zi rows ----------------
__global__ __launch_bounds__(256) void rownorm_kernel(const float* __restrict__ zi,
                                                      const float* __restrict__ zj,
                                                      float* __restrict__ invn,
                                                      float* __restrict__ ws) {
    int row = blockIdx.x;  // 0..2047
    const float* src = (row < B_ROWS) ? (zi + (size_t)row * C_COLS)
                                      : (zj + (size_t)(row - B_ROWS) * C_COLS);
    float sq = 0.0f;
    for (int c = threadIdx.x; c < C_COLS; c += 256) {
        float v = src[c];
        sq += v * v;
    }
    __shared__ float red[256];
    red[threadIdx.x] = sq;
    __syncthreads();
    for (int s = 128; s > 0; s >>= 1) {
        if (threadIdx.x < s) red[threadIdx.x] += red[threadIdx.x + s];
        __syncthreads();
    }
    if (threadIdx.x == 0) {
        float nrm = sqrtf(red[0]);
        invn[row] = 1.0f / fmaxf(nrm, EPS_C);
        if (row < B_ROWS) atomicAdd(&ws[1], nrm);  // z-norm penalty uses zi rows
    }
}

// ---------------- per-column unbiased variance of zi; Σ var over columns ----------------
__global__ __launch_bounds__(256) void colvar_kernel(const float* __restrict__ zi,
                                                     float* __restrict__ ws) {
    int c = blockIdx.x * blockDim.x + threadIdx.x;  // 0..4095
    float s = 0.0f, sq = 0.0f;
    for (int r = 0; r < B_ROWS; ++r) {               // coalesced: tid == column
        float v = zi[(size_t)r * C_COLS + c];
        s += v;
        sq += v * v;
    }
    float var = (sq - s * s * (1.0f / (float)B_ROWS)) * (1.0f / (float)(B_ROWS - 1));
    __shared__ float red[256];
    red[threadIdx.x] = var;
    __syncthreads();
    for (int st = 128; st > 0; st >>= 1) {
        if (threadIdx.x < st) red[threadIdx.x] += red[threadIdx.x + st];
        __syncthreads();
    }
    if (threadIdx.x == 0) atomicAdd(&ws[2], red[0]);
}

// ---------------- Gram via V_WMMA_F32_16X16X32_F16 + fused exp row-sum ----------------
// Raw z staged in f16 (epilogue applies invn_i*invn_j: cosine error ~1.6e-5).
// 128x128 block tile: 8 waves as 4 row-groups x 2 col-groups; each wave computes
// 2 row-tiles x 4 col-tiles of 16x16 -> 16 WMMA per K=64 chunk per wave.
// LDS row stride 72 halves = 144B (16B-aligned, conflict-free bank step).
__global__ __launch_bounds__(256) void gram_kernel(const float* __restrict__ zi,
                                                   const float* __restrict__ zj,
                                                   const float* __restrict__ invn,
                                                   float* __restrict__ rowsum,
                                                   float* __restrict__ pos) {
    __shared__ __attribute__((aligned(16))) _Float16 As[128][72];
    __shared__ __attribute__((aligned(16))) _Float16 Bs[128][72];

    const int tid  = threadIdx.x;
    const int lane = tid & 31;
    const int lm   = lane & 15;    // N within tile / M row select
    const int lh   = lane >> 4;    // half-wave select
    const int wave = tid >> 5;
    const int wrg  = wave & 3;     // row group: rows [32*wrg, 32*wrg+32)
    const int wcg  = wave >> 2;    // col group: cols [64*wcg, 64*wcg+64)
    const int blockM = blockIdx.y * 128;
    const int blockN = blockIdx.x * 128;

    // Tiles never straddle the zi/zj boundary (1024 % 128 == 0): uniform bases.
    const float* baseA = (blockM < B_ROWS) ? (zi + (size_t)blockM * C_COLS)
                                           : (zj + (size_t)(blockM - B_ROWS) * C_COLS);
    const float* baseB = (blockN < B_ROWS) ? (zi + (size_t)blockN * C_COLS)
                                           : (zj + (size_t)(blockN - B_ROWS) * C_COLS);

    v8f acc[2][4] = {};

    for (int kb = 0; kb < C_COLS; kb += 64) {
        // stage A: 128 rows x 64 k = 2048 float4 (8 per thread), cvt f32->f16
#pragma unroll
        for (int it = 0; it < 8; ++it) {
            int idx = tid + it * 256;
            int r = idx >> 4, c4 = idx & 15;
            float4 v = *(const float4*)(baseA + (size_t)r * C_COLS + kb + c4 * 4);
            v4h h = { (_Float16)v.x, (_Float16)v.y, (_Float16)v.z, (_Float16)v.w };
            *(v4h*)&As[r][c4 * 4] = h;
        }
        // stage B: 128 rows x 64 k
#pragma unroll
        for (int it = 0; it < 8; ++it) {
            int idx = tid + it * 256;
            int r = idx >> 4, c4 = idx & 15;
            float4 v = *(const float4*)(baseB + (size_t)r * C_COLS + kb + c4 * 4);
            v4h h = { (_Float16)v.x, (_Float16)v.y, (_Float16)v.z, (_Float16)v.w };
            *(v4h*)&Bs[r][c4 * 4] = h;
        }
        __syncthreads();

#pragma unroll
        for (int k0 = 0; k0 < 64; k0 += 32) {
            v16h a[2], b[4];
            // A 16x32 f16 layout: lane holds K[k0+8*lh..+7] in v0-3, K[k0+16+8*lh..+7] in v4-7
#pragma unroll
            for (int rt = 0; rt < 2; ++rt) {
                const int ar = 32 * wrg + 16 * rt + lm;
                v8h lo = *(const v8h*)&As[ar][k0 + 8 * lh];
                v8h hi = *(const v8h*)&As[ar][k0 + 16 + 8 * lh];
                a[rt] = __builtin_shufflevector(lo, hi, 0,1,2,3,4,5,6,7,8,9,10,11,12,13,14,15);
            }
            // B 32x16 f16 layout: lane (N=lm) holds K[k0+8*lh..+7] in v0-3, K[k0+16+8*lh..+7] in v4-7
#pragma unroll
            for (int ct = 0; ct < 4; ++ct) {
                const int br = 64 * wcg + 16 * ct + lm;
                v8h lo = *(const v8h*)&Bs[br][k0 + 8 * lh];
                v8h hi = *(const v8h*)&Bs[br][k0 + 16 + 8 * lh];
                b[ct] = __builtin_shufflevector(lo, hi, 0,1,2,3,4,5,6,7,8,9,10,11,12,13,14,15);
            }
#pragma unroll
            for (int rt = 0; rt < 2; ++rt)
#pragma unroll
                for (int ct = 0; ct < 4; ++ct)
                    acc[rt][ct] = __builtin_amdgcn_wmma_f32_16x16x32_f16(
                        false, a[rt], false, b[ct], (short)0, acc[rt][ct], false, false);
        }
        __syncthreads();
    }

    // Epilogue: C/D layout -> VGPR r: lanes 0-15 hold M=r, lanes 16-31 hold M=r+8; N=lm.
    float invn_j[4];
#pragma unroll
    for (int ct = 0; ct < 4; ++ct)
        invn_j[ct] = invn[blockN + 64 * wcg + 16 * ct + lm];

#pragma unroll
    for (int rt = 0; rt < 2; ++rt) {
        const int i_base = blockM + 32 * wrg + 16 * rt + 8 * lh;  // multiple of 8
        float4 in0 = *(const float4*)&invn[i_base];
        float4 in1 = *(const float4*)&invn[i_base + 4];
        const float invn_i[8] = { in0.x, in0.y, in0.z, in0.w, in1.x, in1.y, in1.z, in1.w };

#pragma unroll
        for (int r = 0; r < 8; ++r) {
            const int i = i_base + r;
            float esum = 0.0f;
#pragma unroll
            for (int ct = 0; ct < 4; ++ct) {
                const int j = blockN + 64 * wcg + 16 * ct + lm;
                float s = acc[rt][ct][r] * (invn_i[r] * invn_j[ct] * (1.0f / TEMP_C));
                float e = (i == j) ? 0.0f : __expf(s);              // off-diagonal mask
                if (j == i + B_ROWS || i == j + B_ROWS) pos[i] = s; // positive-pair logit
                esum += e;
            }
            float v = esum;  // 4 col-tiles share row i -> one reduction + atomic
            for (int off = 8; off >= 1; off >>= 1) v += __shfl_xor(v, off, 16);
            if (lm == 0) atomicAdd(&rowsum[i], v);
        }
    }
}

// ---------------- finalize: contr = mean(log(rowsum) - pos); combine all terms ----------------
__global__ __launch_bounds__(256) void finalize_kernel(const float* __restrict__ ws,
                                                       float* __restrict__ out) {
    const float* rowsum = ws + 4;
    const float* pos    = ws + 4 + 2048;
    float acc = 0.0f;
    for (int i = threadIdx.x; i < N_ROWS; i += 256)
        acc += __logf(rowsum[i]) - pos[i];
    __shared__ float red[256];
    red[threadIdx.x] = acc;
    __syncthreads();
    for (int s = 128; s > 0; s >>= 1) {
        if (threadIdx.x < s) red[threadIdx.x] += red[threadIdx.x + s];
        __syncthreads();
    }
    if (threadIdx.x == 0) {
        float contr = red[0] * (1.0f / (float)N_ROWS);
        float recon = ws[0] * (1.0f / (float)RECON_N);
        float znorm = ws[1] * (1.0f / (float)B_ROWS);
        float zvar  = ws[2] * (1.0f / (float)C_COLS);
        out[0] = GAMMA_C * contr + (1.0f - GAMMA_C) * recon + ZVP_C * zvar + ZNP_C * znorm;
    }
}

extern "C" void kernel_launch(void* const* d_in, const int* in_sizes, int n_in,
                              void* d_out, int out_size, void* d_ws, size_t ws_size,
                              hipStream_t stream) {
    const float* zi = (const float*)d_in[0];
    const float* zj = (const float*)d_in[1];
    const float* y  = (const float*)d_in[2];
    const float* t  = (const float*)d_in[3];
    const float* m  = (const float*)d_in[4];
    float* ws  = (float*)d_ws;
    float* out = (float*)d_out;

    float* rowsum = ws + 4;
    float* pos    = ws + 4 + 2048;
    float* invn   = ws + 4 + 4096;

    init_ws_kernel<<<(WS_FLOATS + 255) / 256, 256, 0, stream>>>(ws);

    // bandwidth-dominant term: 805 MB streamed once
    recon_kernel<<<2048, 256, 0, stream>>>((const float4*)y, (const float4*)t,
                                           (const float4*)m, ws);

    rownorm_kernel<<<N_ROWS, 256, 0, stream>>>(zi, zj, invn, ws);
    colvar_kernel<<<C_COLS / 256, 256, 0, stream>>>(zi, ws);

    dim3 grid(N_ROWS / 128, N_ROWS / 128);   // (16, 16)
    gram_kernel<<<grid, 256, 0, stream>>>(zi, zj, invn, rowsum, pos);

    finalize_kernel<<<1, 256, 0, stream>>>(ws, out);
}
